// RWKV7_CrossAttBlock_78391743086643
// MI455X (gfx1250) — matrix operand order repaired and checked by reading, not compiled, not run on hardware
//
#include <hip/hip_runtime.h>
#include <math.h>

// ---------------------------------------------------------------------------
// RWKV7 cross-attention block for MI455X (gfx1250, wave32, WMMA).
// GEMMs: bf16 A/B, f32 accumulate via v_wmma_f32_16x16x32_bf16.
// B tiles double-buffered in LDS via global_load_async_to_lds_b128 (ASYNCcnt);
// each wave computes 32x64 (2 M-tiles x 4 N-tiles) reusing B fragments.
// WKV7 recurrence: f32 VALU sequential scan (wrong shape for WMMA).
// ---------------------------------------------------------------------------

typedef __bf16 bh;
typedef __attribute__((ext_vector_type(16))) __bf16 v16bf;
typedef __attribute__((ext_vector_type(8)))  __bf16 v8bf;
typedef __attribute__((ext_vector_type(8)))  float  v8f;

#define BB  2
#define TT  1024
#define CC  1024
#define HH  16
#define NNh 64
#define MT  2048            // B*T rows
#define BTC ((size_t)MT*CC) // elements
#define BSTRIDE 40          // LDS B-tile row stride (elems): 80B -> conflict-free

// ---------------- block reduce (power-of-2 blockDim) ------------------------
__device__ __forceinline__ float bredux(float v, float* sm, int n) {
    int tid = threadIdx.x;
    sm[tid] = v; __syncthreads();
    for (int off = n >> 1; off > 0; off >>= 1) {
        if (tid < off) sm[tid] += sm[tid + off];
        __syncthreads();
    }
    float r = sm[0]; __syncthreads();
    return r;
}

// ---------------- LayerNorm over C=1024, one row per 256-thread block -------
__global__ void ln_k(const float* __restrict__ x, const float* __restrict__ w,
                     const float* __restrict__ b, float* __restrict__ o) {
    __shared__ float sm[256];
    size_t row = blockIdx.x;
    const float* xr = x + row * CC;
    float vals[4]; float s = 0.f, s2 = 0.f;
    #pragma unroll
    for (int i = 0; i < 4; i++) {
        float v = xr[threadIdx.x + i * 256];
        vals[i] = v; s += v; s2 += v * v;
    }
    float sum = bredux(s, sm, 256);
    float ssq = bredux(s2, sm, 256);
    float mu  = sum * (1.f / CC);
    float var = ssq * (1.f / CC) - mu * mu;
    float inv = rsqrtf(var + 1e-5f);
    float* orow = o + row * CC;
    #pragma unroll
    for (int i = 0; i < 4; i++) {
        int c = threadIdx.x + i * 256;
        orow[c] = (vals[i] - mu) * inv * w[c] + b[c];
    }
}

// ---------------- time-shift mix producers (write bf16 for WMMA) ------------
__global__ void mix4_k(const float* __restrict__ x,
                       const float* __restrict__ cr, const float* __restrict__ cw,
                       const float* __restrict__ ca, const float* __restrict__ cg,
                       bh* __restrict__ xr, bh* __restrict__ xw,
                       bh* __restrict__ xa, bh* __restrict__ xg) {
    size_t e = (size_t)blockIdx.x * 256 + threadIdx.x;
    int c = (int)(e & (CC - 1));
    size_t row = e >> 10;
    int t = (int)(row & (TT - 1));
    float xv = x[e];
    float d = (t ? x[e - CC] : 0.f) - xv;
    xr[e] = (bh)(xv + d * cr[c]);
    xw[e] = (bh)(xv + d * cw[c]);
    xa[e] = (bh)(xv + d * ca[c]);
    xg[e] = (bh)(xv + d * cg[c]);
}

__global__ void mix2_k(const float* __restrict__ x,
                       const float* __restrict__ ck, const float* __restrict__ cv,
                       bh* __restrict__ xk, bh* __restrict__ xv) {
    size_t e = (size_t)blockIdx.x * 256 + threadIdx.x;
    int c = (int)(e & (CC - 1));
    size_t row = e >> 10;
    int t = (int)(row & (TT - 1));
    float v0 = x[e];
    float d = (t ? x[e - CC] : 0.f) - v0;
    xk[e] = (bh)(v0 + d * ck[c]);
    xv[e] = (bh)(v0 + d * cv[c]);
}

__global__ void mix1_k(const float* __restrict__ x, const float* __restrict__ ck,
                       bh* __restrict__ xk) {
    size_t e = (size_t)blockIdx.x * 256 + threadIdx.x;
    int c = (int)(e & (CC - 1));
    size_t row = e >> 10;
    int t = (int)(row & (TT - 1));
    float v0 = x[e];
    float d = (t ? x[e - CC] : 0.f) - v0;
    xk[e] = (bh)(v0 + d * ck[c]);
}

// ---------------- transpose + f32->bf16 weight conversion -------------------
// W is [K,N] row-major f32; Wt is [N,K] row-major bf16 (B-fragment friendly).
__global__ void tr_cvt(const float* __restrict__ W, bh* __restrict__ Wt,
                       int K, int N) {
    __shared__ float tile[32][33];
    int kb = blockIdx.y * 32, nb = blockIdx.x * 32;
    for (int i = threadIdx.y; i < 32; i += 8)
        tile[i][threadIdx.x] = W[(size_t)(kb + i) * N + nb + threadIdx.x];
    __syncthreads();
    for (int i = threadIdx.y; i < 32; i += 8)
        Wt[(size_t)(nb + i) * K + kb + threadIdx.x] = (bh)tile[threadIdx.x][i];
}

// ---------------- WMMA GEMM: out[M,N] = A[M,K] @ Wt[N,K]^T ------------------
// 256 threads = 8 waves; wave computes a 32x64 strip: 2 M-tiles x 4 N-tiles,
// 8 WMMAs per k-step reusing 4 LDS B fragments. The 64x32 B k-tile is shared
// by all 8 waves: staged into LDS once per block with async copies (ASYNCcnt)
// and double-buffered against the WMMAs.
// epi: 0 = f32 out (+optional residual add), 1 = tanh->bf16, 2 = sigmoid->bf16,
//      3 = relu^2 -> bf16, 4 = plain ->bf16.
__global__ void gemm_wmma(const bh* __restrict__ A, const bh* __restrict__ Wt,
                          float* __restrict__ outf, bh* __restrict__ outb,
                          const float* __restrict__ addsrc,
                          int N, int K, int epi) {
    __shared__ __align__(16) bh Bs[2][64 * BSTRIDE];

    int tid   = threadIdx.x;
    int lane  = tid & 31;
    int wave  = tid >> 5;
    int row0  = blockIdx.y * 256 + wave * 32;
    int col0  = blockIdx.x * 64;
    int lrow  = lane & 15;
    int ahalf = (lane >> 4) * 8;    // A: K sub-block select per ISA layout
    int bhalf = (lane >> 4) * 8;    // B: K sub-block select per ISA layout

    // Async staging assignment: thread tid copies one 16B chunk of the tile.
    int scol   = tid >> 2;          // 0..63  (column of B tile)
    int schunk = (tid & 3) * 8;     // 0,8,16,24 (k offset, elems; 16B chunks)
    const bh* bgp = Wt + (size_t)(col0 + scol) * K + schunk;
    unsigned lds0 = (unsigned)(size_t)(&Bs[0][scol * BSTRIDE + schunk]);
    unsigned lds1 = (unsigned)(size_t)(&Bs[1][scol * BSTRIDE + schunk]);

    // Prologue: stage k-tile 0 into buffer 0.
    asm volatile("global_load_async_to_lds_b128 %0, %1, off"
                 :: "v"(lds0), "v"((unsigned long long)bgp) : "memory");
    asm volatile("s_wait_asynccnt 0" ::: "memory");
    __syncthreads();

    v8f acc0[4] = {};
    v8f acc1[4] = {};
    const bh* ap0 = A + (size_t)(row0 + lrow) * K + ahalf;
    const bh* ap1 = ap0 + (size_t)16 * K;

    int nk = K >> 5;
    for (int kt = 0; kt < nk; kt++) {
        int buf = kt & 1;
        // Stage next k-tile into the other buffer (overlaps with WMMAs below).
        if (kt + 1 < nk) {
            unsigned long long bgn = (unsigned long long)(bgp + (kt + 1) * 32);
            unsigned ldsn = buf ? lds0 : lds1;
            asm volatile("global_load_async_to_lds_b128 %0, %1, off"
                         :: "v"(ldsn), "v"(bgn) : "memory");
        }
        // Two A fragments: per-wave private, direct from global.
        union { v16bf v; v8bf h[2]; } af0, af1;
        af0.h[0] = *(const v8bf*)(ap0);
        af0.h[1] = *(const v8bf*)(ap0 + 16);
        af1.h[0] = *(const v8bf*)(ap1);
        af1.h[1] = *(const v8bf*)(ap1 + 16);
        __builtin_prefetch(ap0 + 64, 0, 1);
        ap0 += 32; ap1 += 32;
        // All 4 B fragments up-front (one ds clause, latency hidden by WMMAs).
        union { v16bf v; v8bf h[2]; } bfr[4];
        #pragma unroll
        for (int t = 0; t < 4; t++) {
            const bh* bp = &Bs[buf][(t * 16 + lrow) * BSTRIDE + bhalf];
            bfr[t].h[0] = *(const v8bf*)(bp);
            bfr[t].h[1] = *(const v8bf*)(bp + 16);
        }
        // 8 WMMAs reuse the 4 B fragments across 2 M-tiles.
        #pragma unroll
        for (int t = 0; t < 4; t++) {
            acc0[t] = __builtin_amdgcn_wmma_f32_16x16x32_bf16(
                false, af0.v, false, bfr[t].v, (short)0, acc0[t], false, false);
            acc1[t] = __builtin_amdgcn_wmma_f32_16x16x32_bf16(
                false, af1.v, false, bfr[t].v, (short)0, acc1[t], false, false);
        }
        // Next tile must have landed; all waves done reading current buffer.
        if (kt + 1 < nk) {
            asm volatile("s_wait_asynccnt 0" ::: "memory");
        }
        __syncthreads();
    }

    int crow = row0 + (lane >> 4) * 8;
    #pragma unroll
    for (int t = 0; t < 4; t++) {
        int ccol = col0 + t * 16 + lrow;
        #pragma unroll
        for (int i = 0; i < 8; i++) {
            #pragma unroll
            for (int m = 0; m < 2; m++) {
                size_t idx = (size_t)(crow + m * 16 + i) * N + ccol;
                float val = m ? acc1[t][i] : acc0[t][i];
                if (addsrc) val += addsrc[idx];
                if (epi == 0)      outf[idx] = val;
                else if (epi == 1) outb[idx] = (bh)tanhf(val);
                else if (epi == 2) outb[idx] = (bh)(1.f / (1.f + expf(-val)));
                else if (epi == 3) { float rl = fmaxf(val, 0.f); outb[idx] = (bh)(rl * rl); }
                else               outb[idx] = (bh)val;
            }
        }
    }
}

// ---------------- WKV7 input prep (decay, a, normalized kk, k') -------------
__global__ void wkv_prep(const float* __restrict__ k, const float* __restrict__ t2,
                         const float* __restrict__ a2t,
                         const float* __restrict__ w0, const float* __restrict__ a0,
                         const float* __restrict__ k_k, const float* __restrict__ k_a,
                         float* __restrict__ dec, float* __restrict__ kp,
                         float* __restrict__ aa, float* __restrict__ bb) {
    __shared__ float sm[64];
    int tid = threadIdx.x;
    size_t bth = blockIdx.x;           // (b*T+t)*H + h
    int h = (int)(bth & (HH - 1));
    size_t e = bth * NNh + tid;
    int c = (h << 6) + tid;
    float kv = k[e];
    float wpre = w0[c] + t2[e];
    float wraw = -log1pf(expf(-wpre)) - 0.5f;      // -softplus(-x) - 0.5
    float decv = expf(-expf(wraw));
    float av = 1.f / (1.f + expf(-(a0[c] + a2t[e])));
    float kk = kv * k_k[c];
    float ss = bredux(kk * kk, sm, 64);
    float kkn = kk / fmaxf(sqrtf(ss), 1e-12f);
    dec[e] = decv;
    aa[e] = -kkn;
    bb[e] = kkn * av;
    kp[e] = kv * (1.f + (av - 1.f) * k_a[c]);
}

// ---------------- WKV7 sequential scan: one block per (b,h) -----------------
// Thread i owns state row S[i][:], 64 VGPRs of f32.
__global__ void wkv_scan(const float* __restrict__ r, const float* __restrict__ w,
                         const float* __restrict__ k, const float* __restrict__ v,
                         const float* __restrict__ a, const float* __restrict__ b,
                         float* __restrict__ y) {
    int i = threadIdx.x;
    int bh_ = blockIdx.x;
    int bat = bh_ >> 4, h = bh_ & (HH - 1);
    float S[NNh];
    #pragma unroll
    for (int j = 0; j < NNh; j++) S[j] = 0.f;
    __shared__ float sr[64], sw[64], sk[64], sa[64], sb[64];
    for (int t = 0; t < TT; t++) {
        size_t base = (((size_t)bat * TT + t) * HH + h) * NNh;
        sr[i] = r[base + i]; sw[i] = w[base + i]; sk[i] = k[base + i];
        sa[i] = a[base + i]; sb[i] = b[base + i];
        float vi = v[base + i];
        __syncthreads();
        float sacc = 0.f;
        #pragma unroll
        for (int j = 0; j < NNh; j++) sacc += S[j] * sa[j];
        float yv = 0.f;
        #pragma unroll
        for (int j = 0; j < NNh; j++) {
            S[j] = S[j] * sw[j] + sacc * sb[j] + vi * sk[j];
            yv += S[j] * sr[j];
        }
        y[base + i] = yv;
        __syncthreads();
    }
}

// ---------------- per-head GroupNorm + bonus + gate -> bf16 -----------------
__global__ void post_k(const float* __restrict__ y, const float* __restrict__ r,
                       const float* __restrict__ k, const float* __restrict__ v,
                       const float* __restrict__ g, const float* __restrict__ r_k,
                       const float* __restrict__ ln_w, const float* __restrict__ ln_b,
                       bh* __restrict__ out) {
    __shared__ float sm[64];
    int i = threadIdx.x;
    size_t bth = blockIdx.x;
    int h = (int)(bth & (HH - 1));
    size_t e = bth * NNh + i;
    int c = (h << 6) + i;
    float yv = y[e];
    float mu = bredux(yv, sm, 64) * (1.f / NNh);
    float d = yv - mu;
    float var = bredux(d * d, sm, 64) * (1.f / NNh);
    float nrm = d * rsqrtf(var + 64e-5f) * ln_w[c] + ln_b[c];
    float dot = bredux(r[e] * k[e] * r_k[c], sm, 64);
    out[e] = (bh)((nrm + dot * v[e]) * g[e]);
}

// ===========================================================================
// Host-side orchestration
// ===========================================================================

struct Ctx {
    hipStream_t s;
    // f32 activations
    float *x, *x2, *ln, *kv;
    float *r_f, *k_f, *v_f, *t2, *a2t, *gbuf, *dec, *kp, *aaq, *bbq, *ybuf;
    // bf16 activations
    bh *xr, *xw, *xa, *xg, *xkb, *xvb, *ybf, *t1b, *a1b, *g1b, *hbuf, *fxk;
    // bf16 transposed weights (reused per stage)
    bh *Wrt, *Wkt, *Wvt, *Wot, *w1t, *w2t, *a1t, *a2tw, *g1t, *g2t, *wfk, *wfv;
};

static void launch_ln(hipStream_t s, const float* x, const float* w,
                      const float* b, float* o) {
    ln_k<<<MT, 256, 0, s>>>(x, w, b, o);
}
static void launch_tr(hipStream_t s, const float* W, bh* Wt, int K, int N) {
    tr_cvt<<<dim3(N / 32, K / 32), dim3(32, 8), 0, s>>>(W, Wt, K, N);
}
static void launch_gemm(hipStream_t s, const bh* A, const bh* Wt, float* of,
                        bh* ob, const float* add, int N, int K, int epi) {
    gemm_wmma<<<dim3(N / 64, MT / 256), 256, 0, s>>>(A, Wt, of, ob, add, N, K, epi);
}

// tmix param indices: 0 x_r, 1 x_w, 2 x_k, 3 x_v, 4 x_a, 5 x_g, 6 Wr, 7 Wk,
// 8 Wv, 9 Wo, 10 w0, 11 w1, 12 w2, 13 a0, 14 a1, 15 a2, 16 g1, 17 g2,
// 18 k_k, 19 k_a, 20 r_k, 21 ln_w, 22 ln_b
static void run_tmix(Ctx& c, const float* xq, const float* xkv,
                     const float* const* P, const float* addsrc,
                     float* outx, float* vdst) {
    hipStream_t s = c.s;
    mix4_k<<<BTC / 256, 256, 0, s>>>(xq, P[0], P[1], P[4], P[5], c.xr, c.xw, c.xa, c.xg);
    mix2_k<<<BTC / 256, 256, 0, s>>>(xkv, P[2], P[3], c.xkb, c.xvb);
    launch_tr(s, P[6],  c.Wrt, CC, CC);
    launch_tr(s, P[7],  c.Wkt, CC, CC);
    launch_tr(s, P[8],  c.Wvt, CC, CC);
    launch_tr(s, P[9],  c.Wot, CC, CC);
    launch_tr(s, P[11], c.w1t, CC, 64);
    launch_tr(s, P[12], c.w2t, 64, CC);
    launch_tr(s, P[14], c.a1t, CC, 64);
    launch_tr(s, P[15], c.a2tw, 64, CC);
    launch_tr(s, P[16], c.g1t, CC, 128);
    launch_tr(s, P[17], c.g2t, 128, CC);
    // big projections
    launch_gemm(s, c.xr,  c.Wrt, c.r_f, nullptr, nullptr, CC, CC, 0);
    launch_gemm(s, c.xkb, c.Wkt, c.k_f, nullptr, nullptr, CC, CC, 0);
    launch_gemm(s, c.xvb, c.Wvt, c.v_f, nullptr, nullptr, CC, CC, 0);
    // decay LoRA: tanh(xw@w1)@w2
    launch_gemm(s, c.xw, c.w1t, nullptr, c.t1b, nullptr, 64, CC, 1);
    launch_gemm(s, c.t1b, c.w2t, c.t2, nullptr, nullptr, CC, 64, 0);
    // a LoRA: (xa@a1)@a2
    launch_gemm(s, c.xa, c.a1t, nullptr, c.a1b, nullptr, 64, CC, 4);
    launch_gemm(s, c.a1b, c.a2tw, c.a2t, nullptr, nullptr, CC, 64, 0);
    // gate: sigmoid(xg@g1)@g2
    launch_gemm(s, c.xg, c.g1t, nullptr, c.g1b, nullptr, 128, CC, 2);
    launch_gemm(s, c.g1b, c.g2t, c.gbuf, nullptr, nullptr, CC, 128, 0);
    // WKV7
    wkv_prep<<<BB * TT * HH, 64, 0, s>>>(c.k_f, c.t2, c.a2t, P[10], P[13],
                                         P[18], P[19], c.dec, c.kp, c.aaq, c.bbq);
    wkv_scan<<<BB * HH, 64, 0, s>>>(c.r_f, c.dec, c.kp, c.v_f, c.aaq, c.bbq, c.ybuf);
    post_k<<<BB * TT * HH, 64, 0, s>>>(c.ybuf, c.r_f, c.kp, c.v_f, c.gbuf,
                                       P[20], P[21], P[22], c.ybf);
    // output projection with fused residual add
    launch_gemm(s, c.ybf, c.Wot, outx, nullptr, addsrc, CC, CC, 0);
    // v_first output
    hipMemcpyAsync(vdst, c.v_f, BTC * sizeof(float), hipMemcpyDeviceToDevice, s);
}

static void run_ffn(Ctx& c, const float* xin, const float* xk_coef,
                    const float* Wk, const float* Wv,
                    const float* addsrc, float* out) {
    hipStream_t s = c.s;
    mix1_k<<<BTC / 256, 256, 0, s>>>(xin, xk_coef, c.fxk);
    launch_tr(s, Wk, c.wfk, CC, 4 * CC);
    launch_tr(s, Wv, c.wfv, 4 * CC, CC);
    launch_gemm(s, c.fxk, c.wfk, nullptr, c.hbuf, nullptr, 4 * CC, CC, 3);
    launch_gemm(s, c.hbuf, c.wfv, out, nullptr, addsrc, CC, 4 * CC, 0);
}

extern "C" void kernel_launch(void* const* d_in, const int* in_sizes, int n_in,
                              void* d_out, int out_size, void* d_ws, size_t ws_size,
                              hipStream_t stream) {
    (void)in_sizes; (void)n_in; (void)out_size; (void)ws_size;
    const float* query  = (const float*)d_in[0];
    const float* keyval = (const float*)d_in[1];
    // d_in[2], d_in[3] (incoming v_first) unused: layer_idx==0 overwrites them.
    int ip = 4;
    const float *lnw[7], *lnb[7]; // sa_pre, sa_attn, sa_ffn_n, pre, q_n, kv_n, ffn_n
    for (int i = 0; i < 7; i++) { lnw[i] = (const float*)d_in[ip++]; lnb[i] = (const float*)d_in[ip++]; }
    const float* saP[23]; for (int i = 0; i < 23; i++) saP[i] = (const float*)d_in[ip++];
    const float* saffn_xk = (const float*)d_in[ip++];
    const float* saffn_Wk = (const float*)d_in[ip++];
    const float* saffn_Wv = (const float*)d_in[ip++];
    const float* caP[23]; for (int i = 0; i < 23; i++) caP[i] = (const float*)d_in[ip++];
    const float* ffn_xk = (const float*)d_in[ip++];
    const float* ffn_Wk = (const float*)d_in[ip++];
    const float* ffn_Wv = (const float*)d_in[ip++];

    // ---- workspace bump allocator ----
    char* wp = (char*)d_ws;
    auto alloc = [&](size_t bytes) -> void* {
        void* r = (void*)wp;
        wp += (bytes + 255) & ~(size_t)255;
        return r;
    };
    Ctx c; c.s = stream;
    const size_t F4 = (size_t)BTC * 4;      // f32 [B,T,C]
    const size_t H2 = (size_t)BTC * 2;      // bf16 [B,T,C]
    c.x   = (float*)alloc(F4);  c.x2 = (float*)alloc(F4);
    c.ln  = (float*)alloc(F4);  c.kv = (float*)alloc(F4);
    c.r_f = (float*)alloc(F4);  c.k_f = (float*)alloc(F4);  c.v_f = (float*)alloc(F4);
    c.t2  = (float*)alloc(F4);  c.a2t = (float*)alloc(F4);  c.gbuf = (float*)alloc(F4);
    c.dec = (float*)alloc(F4);  c.kp  = (float*)alloc(F4);
    c.aaq = (float*)alloc(F4);  c.bbq = (float*)alloc(F4);  c.ybuf = (float*)alloc(F4);
    c.xr  = (bh*)alloc(H2); c.xw = (bh*)alloc(H2); c.xa = (bh*)alloc(H2);
    c.xg  = (bh*)alloc(H2); c.xkb = (bh*)alloc(H2); c.xvb = (bh*)alloc(H2);
    c.ybf = (bh*)alloc(H2); c.fxk = (bh*)alloc(H2);
    c.t1b = (bh*)alloc((size_t)MT * 64 * 2);
    c.a1b = (bh*)alloc((size_t)MT * 64 * 2);
    c.g1b = (bh*)alloc((size_t)MT * 128 * 2);
    c.hbuf = (bh*)alloc((size_t)MT * 4 * CC * 2);
    c.Wrt = (bh*)alloc((size_t)CC * CC * 2); c.Wkt = (bh*)alloc((size_t)CC * CC * 2);
    c.Wvt = (bh*)alloc((size_t)CC * CC * 2); c.Wot = (bh*)alloc((size_t)CC * CC * 2);
    c.w1t = (bh*)alloc((size_t)64 * CC * 2);  c.w2t = (bh*)alloc((size_t)CC * 64 * 2);
    c.a1t = (bh*)alloc((size_t)64 * CC * 2);  c.a2tw = (bh*)alloc((size_t)CC * 64 * 2);
    c.g1t = (bh*)alloc((size_t)128 * CC * 2); c.g2t = (bh*)alloc((size_t)CC * 128 * 2);
    c.wfk = (bh*)alloc((size_t)4 * CC * CC * 2);
    c.wfv = (bh*)alloc((size_t)4 * CC * CC * 2);

    float* outq  = (float*)d_out;
    float* outv1 = outq + BTC;
    float* outv2 = outq + 2 * BTC;

    // ---- inner self-attn RWKV7 block ----
    launch_ln(stream, query, lnw[0], lnb[0], c.x);        // x = LN(query, sa_pre)
    launch_ln(stream, c.x, lnw[1], lnb[1], c.ln);         // xn = LN(x, sa_attn)
    run_tmix(c, c.ln, c.ln, saP, /*add=*/c.x, /*out=*/c.x2, /*v->*/outv1);
    launch_ln(stream, c.x2, lnw[2], lnb[2], c.ln);        // LN(x, sa_ffn_n)
    run_ffn(c, c.ln, saffn_xk, saffn_Wk, saffn_Wv, /*add=*/c.x2, /*out=*/c.x);

    // ---- outer cross-attn ----
    launch_ln(stream, c.x, lnw[3], lnb[3], c.x2);         // q = LN(x, pre)
    launch_ln(stream, c.x2, lnw[4], lnb[4], c.ln);        // qn = LN(q, q_n)
    launch_ln(stream, keyval, lnw[5], lnb[5], c.kv);      // kvn = LN(keyval, kv_n)
    run_tmix(c, c.ln, c.kv, caP, /*add=*/c.x2, /*out=*/c.x, /*v->*/outv2);
    launch_ln(stream, c.x, lnw[6], lnb[6], c.ln);         // LN(q, ffn_n)
    run_ffn(c, c.ln, ffn_xk, ffn_Wk, ffn_Wv, /*add=*/c.x, /*out=*/c.x2);

    hipMemcpyAsync(outq, c.x2, BTC * sizeof(float), hipMemcpyDeviceToDevice, stream);
}
